// LSTM_49211735277961
// MI455X (gfx1250) — compile-verified, matrix operand order checked by reading, not run on hardware
//
#include <hip/hip_runtime.h>
#include <hip/hip_bf16.h>

typedef __attribute__((ext_vector_type(16))) __bf16 v16bf;
typedef __attribute__((ext_vector_type(8)))  __bf16 v8bf;
typedef __attribute__((ext_vector_type(4)))  __bf16 v4bf;
typedef __attribute__((ext_vector_type(8)))  float  v8f;

#define LSTM_B 128
#define LSTM_T 1024
#define LSTM_D 512
#define LSTM_H 512
#define LSTM_G 2048   // 4*H
#define NTILES_G 128  // G/16
#define SCAN_BLOCKS 32

__device__ __forceinline__ v16bf mkfrag(v8bf lo, v8bf hi) {
    return __builtin_shufflevector(lo, hi, 0,1,2,3,4,5,6,7,8,9,10,11,12,13,14,15);
}

__device__ __forceinline__ float sigmoid_fast(float x) {
    return 1.0f / (1.0f + __expf(-x));
}
__device__ __forceinline__ float tanh_fast(float x) {
    return 1.0f - 2.0f / (__expf(2.0f * x) + 1.0f);
}

// ---------------------------------------------------------------------------
// f32 -> bf16 bulk convert (x), 4 elements/thread.
// ---------------------------------------------------------------------------
__global__ __launch_bounds__(256) void cvt_f32_to_bf16(
    const float* __restrict__ src, __bf16* __restrict__ dst, int n4) {
    int i = blockIdx.x * blockDim.x + threadIdx.x;
    if (i >= n4) return;
    const float4 v = ((const float4*)src)[i];
    v4bf o; o[0] = (__bf16)v.x; o[1] = (__bf16)v.y; o[2] = (__bf16)v.z; o[3] = (__bf16)v.w;
    *(v4bf*)(dst + (size_t)i * 4) = o;
}

// ---------------------------------------------------------------------------
// Pack (K x N) f32 weights into WMMA-B-fragment order (ISA 7.12.2), bf16:
//   flat = (((ktile*NT + ntile)*2 + g)*32 + lane)*8 + s
//   src  k = ktile*32 + g*16 + (lane>>4)*8 + s ; col = ntile*16 + (lane&15)
// Every B-fragment load becomes two contiguous 16B reads per lane.
// ---------------------------------------------------------------------------
__global__ __launch_bounds__(256) void pack_weights(
    const float* __restrict__ W, __bf16* __restrict__ out, int N, int total) {
    int idx = blockIdx.x * blockDim.x + threadIdx.x;
    if (idx >= total) return;
    const int s     = idx & 7;
    const int lane  = (idx >> 3) & 31;
    const int g     = (idx >> 8) & 1;
    const int t3    = idx >> 9;
    const int ntile = t3 & (NTILES_G - 1);
    const int ktile = t3 >> 7;
    const int k   = ktile * 32 + g * 16 + (lane >> 4) * 8 + s;
    const int col = ntile * 16 + (lane & 15);
    out[idx] = (__bf16)W[(size_t)k * N + col];
}

__global__ __launch_bounds__(256) void init_state(
    __bf16* __restrict__ h0, unsigned* __restrict__ bar, int n) {
    int i = blockIdx.x * blockDim.x + threadIdx.x;
    if (i >= n) return;
    h0[i] = (__bf16)0.0f;
    if (i == 0) *bar = 0u;
}

// ---------------------------------------------------------------------------
// Phase 1 GEMM: xg = x_bf16 (M x K) @ Wx_packed + bias, f32 out.
// 8 waves/block arranged 2(M) x 4(N); each wave: 32 x 64 output
// (2 A-fragments x 4 B-fragments -> 8 WMMAs per 32-deep k-step).
// Lane pointers advance by constants; tile selection via immediate offsets.
// ---------------------------------------------------------------------------
__global__ __launch_bounds__(256) void gemm_bf16_wmma(
    const __bf16* __restrict__ X,    // (M, K) row-major bf16
    const __bf16* __restrict__ Wp,   // packed fragments
    const float*  __restrict__ bias, // (N)
    float* __restrict__ Y,           // (M, N)
    int M, int N, int K) {
    const int lane = threadIdx.x & 31;
    const int wave = threadIdx.x >> 5;
    const int m0 = blockIdx.y * 64 + (wave >> 2) * 32;
    const int n0 = blockIdx.x * 256 + (wave & 3) * 64;
    const int kstride_b = (N >> 4) * 512;     // packed elements per k-step

    const __bf16* pa = X + (size_t)(m0 + (lane & 15)) * K + (lane >> 4) * 8;
    const __bf16* pb = Wp + (size_t)(n0 >> 4) * 512 + lane * 8;

    v8f acc[2][4];
#pragma unroll
    for (int i = 0; i < 2; ++i)
#pragma unroll
        for (int j = 0; j < 4; ++j) acc[i][j] = v8f{};

    for (int kt = 0; kt < (K >> 5); ++kt) {
        v16bf a0 = mkfrag(*(const v8bf*)(pa),              *(const v8bf*)(pa + 16));
        v16bf a1 = mkfrag(*(const v8bf*)(pa + 16 * K),     *(const v8bf*)(pa + 16 * K + 16));
#pragma unroll
        for (int j = 0; j < 4; ++j) {
            v16bf b = mkfrag(*(const v8bf*)(pb + j * 512),
                             *(const v8bf*)(pb + j * 512 + 256));
            acc[0][j] = __builtin_amdgcn_wmma_f32_16x16x32_bf16(
                false, a0, false, b, (short)0, acc[0][j], false, false);
            acc[1][j] = __builtin_amdgcn_wmma_f32_16x16x32_bf16(
                false, a1, false, b, (short)0, acc[1][j], false, false);
        }
        pa += 32;
        pb += kstride_b;
    }
    // C/D layout: elem r -> row +8*(lane>>4)+r, col (lane&15)
    const int rbase = (lane >> 4) * 8;
    const int col   = lane & 15;
#pragma unroll
    for (int i = 0; i < 2; ++i) {
#pragma unroll
        for (int j = 0; j < 4; ++j) {
            const int n = n0 + j * 16 + col;
            const float bn = bias[n];
#pragma unroll
            for (int r = 0; r < 8; ++r) {
                const int m = m0 + i * 16 + rbase + r;
                Y[(size_t)m * N + n] = acc[i][j][r] + bn;
            }
        }
    }
}

// ---------------------------------------------------------------------------
// Phase 2: persistent scan kernel. Grid = 32 blocks x 256 threads.
// Block owns hidden tile nt = blockIdx.x (16 cols of H across all 4 gates);
// wave w owns batch rows [16w, 16w+16). Its Wh working set (64 tiles = 64KB)
// is staged into LDS once; B-fragments then come from ds_load_b128 for all
// 1024 timesteps. Cell state c lives in 8 VGPRs/lane for the entire scan.
// Cross-step dependency enforced by a device-scope atomic grid barrier.
// ---------------------------------------------------------------------------
__global__ __launch_bounds__(256) void lstm_scan(
    const float*  __restrict__ xg,    // (T, 128, 2048)
    const __bf16* __restrict__ Whp,   // packed fragments (K=512, N=2048)
    __bf16* hA, __bf16* hB,           // (128, 512) bf16 double buffer
    float* __restrict__ Hout,         // (128, 512) f32 final output
    unsigned* bar) {
    __shared__ __attribute__((aligned(16))) __bf16 wlds[64 * 512];  // 64 KB

    const int lane = threadIdx.x & 31;
    const int wave = threadIdx.x >> 5;
    const int nt   = blockIdx.x;      // hidden tile 0..31
    const int m0   = wave * 16;       // batch rows

    // Stage this block's Wh tiles: (kt 0..15) x (gi 0..3) -> lds tile kt*4+gi.
    for (int c = threadIdx.x; c < 4096; c += 256) {       // 16B chunks
        const int tile  = c >> 6;                          // 0..63
        const int chunk = c & 63;
        const int kt = tile >> 2, gi = tile & 3;
        ((v8bf*)wlds)[tile * 64 + chunk] =
            ((const v8bf*)(Whp + (size_t)(kt * NTILES_G + gi * 32 + nt) * 512))[chunk];
    }
    __syncthreads();

    const int rbase = (lane >> 4) * 8;
    const int col   = nt * 16 + (lane & 15);
    const __bf16* lb = wlds + lane * 8;     // LDS lane pointer

    float creg[8];
#pragma unroll
    for (int r = 0; r < 8; ++r) creg[r] = 0.0f;

    for (int t = 0; t < LSTM_T; ++t) {
        const __bf16* hin  = (t & 1) ? hB : hA;
        __bf16*       hout = (t & 1) ? hA : hB;
        const float* xg_t = xg + (size_t)t * LSTM_B * LSTM_G;

        const __bf16* pa = hin + (size_t)(m0 + (lane & 15)) * LSTM_H + (lane >> 4) * 8;

        v8f acc[4] = {v8f{}, v8f{}, v8f{}, v8f{}};
        const __bf16* lbk = lb;
        for (int kt = 0; kt < 16; ++kt) {
            v16bf a = mkfrag(*(const v8bf*)(pa), *(const v8bf*)(pa + 16));
#pragma unroll
            for (int gi = 0; gi < 4; ++gi) {
                v16bf b = mkfrag(*(const v8bf*)(lbk + gi * 512),
                                 *(const v8bf*)(lbk + gi * 512 + 256));
                acc[gi] = __builtin_amdgcn_wmma_f32_16x16x32_bf16(
                    false, a, false, b, (short)0, acc[gi], false, false);
            }
            pa  += 32;
            lbk += 4 * 512;
        }

#pragma unroll
        for (int r = 0; r < 8; ++r) {
            const int m = m0 + rbase + r;
            const size_t gb = (size_t)m * LSTM_G + col;
            const float iv = sigmoid_fast(acc[0][r] + xg_t[gb]);
            const float fv = sigmoid_fast(acc[1][r] + xg_t[gb + 512]);
            const float gv = tanh_fast   (acc[2][r] + xg_t[gb + 1024]);
            const float ov = sigmoid_fast(acc[3][r] + xg_t[gb + 1536]);
            const float cn = fv * creg[r] + iv * gv;
            creg[r] = cn;
            const float hn = ov * tanh_fast(cn);
            const size_t idx = (size_t)m * LSTM_H + col;
            hout[idx] = (__bf16)hn;
            if (t == LSTM_T - 1) Hout[idx] = hn;
        }

        // --- device-wide step barrier (release h writes, acquire for reads) ---
        __threadfence();                    // release: drain this wave's stores
        __syncthreads();                    // whole block done
        if (threadIdx.x == 0) {
            __hip_atomic_fetch_add(bar, 1u, __ATOMIC_RELEASE,
                                   __HIP_MEMORY_SCOPE_AGENT);
            const unsigned target = (unsigned)SCAN_BLOCKS * (unsigned)(t + 1);
            while (__hip_atomic_load(bar, __ATOMIC_ACQUIRE,
                                     __HIP_MEMORY_SCOPE_AGENT) < target)
                __builtin_amdgcn_s_sleep(2);
        }
        __syncthreads();
        __threadfence();                    // acquire: invalidate stale h lines
    }
}

// ---------------------------------------------------------------------------
extern "C" void kernel_launch(void* const* d_in, const int* in_sizes, int n_in,
                              void* d_out, int out_size, void* d_ws, size_t ws_size,
                              hipStream_t stream) {
    (void)in_sizes; (void)n_in; (void)out_size; (void)ws_size;
    const float* x  = (const float*)d_in[0];  // (B, T, D)
    const float* Wx = (const float*)d_in[1];  // (D, 4H)
    const float* Wh = (const float*)d_in[2];  // (H, 4H)
    const float* b  = (const float*)d_in[3];  // (4H)
    float* out = (float*)d_out;               // (B, H)

    char* ws = (char*)d_ws;
    size_t off = 0;
    float*  xg  = (float*)(ws + off);  off += (size_t)LSTM_B * LSTM_T * LSTM_G * sizeof(float);
    __bf16* xbf = (__bf16*)(ws + off); off += (size_t)LSTM_B * LSTM_T * LSTM_D * sizeof(__bf16);
    __bf16* wxp = (__bf16*)(ws + off); off += (size_t)LSTM_D * LSTM_G * sizeof(__bf16);
    __bf16* whp = (__bf16*)(ws + off); off += (size_t)LSTM_H * LSTM_G * sizeof(__bf16);
    __bf16* hA  = (__bf16*)(ws + off); off += (size_t)LSTM_B * LSTM_H * sizeof(__bf16);
    __bf16* hB  = (__bf16*)(ws + off); off += (size_t)LSTM_B * LSTM_H * sizeof(__bf16);
    unsigned* bar = (unsigned*)(ws + off); off += 256;

    // Phase 0: conversion + weight packing + state/barrier init.
    {
        const int n4 = (LSTM_B * LSTM_T * LSTM_D) / 4;
        cvt_f32_to_bf16<<<n4 / 256, 256, 0, stream>>>(x, xbf, n4);
        const int wtot = LSTM_D * LSTM_G;
        pack_weights<<<wtot / 256, 256, 0, stream>>>(Wx, wxp, LSTM_G, wtot);
        pack_weights<<<wtot / 256, 256, 0, stream>>>(Wh, whp, LSTM_G, wtot);
        const int st = LSTM_B * LSTM_H;
        init_state<<<st / 256, 256, 0, stream>>>(hA, bar, st);
    }

    // Phase 1: xg = x @ Wx + b  (M = 131072, N = 2048, K = 512)
    {
        dim3 grid(LSTM_G / 256, (LSTM_B * LSTM_T) / 64);
        gemm_bf16_wmma<<<grid, 256, 0, stream>>>(
            xbf, wxp, b, xg, LSTM_B * LSTM_T, LSTM_G, LSTM_D);
    }

    // Phase 2: single persistent kernel runs all 1024 timesteps with an
    // atomic grid barrier per step (no per-step launch latency).
    lstm_scan<<<SCAN_BLOCKS, 256, 0, stream>>>(xg, whp, hA, hB, out, bar);
}
